// DigitConvolutionalModel_36919538877097
// MI455X (gfx1250) — matrix operand order repaired and checked by reading, not compiled, not run on hardware
//
#include <hip/hip_runtime.h>

#define BATCH        65536
#define IMGD         28
#define CONV_OUT     676     // 26*26
#define ROWS_PER_BLK 64
#define THREADS      256

// LDS layout (bytes)
#define FEAT_OFF 0            // bf16 [64][704]   (K 676 padded to 704)
#define H1_OFF   90112        // bf16 [64][320]   (K 300 padded to 320)
#define H2_OFF   131072       // bf16 [64][128]   (K 100 padded to 128)
#define XS_OFF   90112        // float[16][784]   (aliases h1/h2 during conv)
#define SMEM_BYTES 147456

// Packed-B geometry (tiles of 32x16 bf16 = 512 elements, lane-layout order)
#define NT1 19
#define KT1 22
#define NT2 7
#define KT2 10
#define NT3 1
#define KT3 4
#define PK_OFF1 0
#define PK_OFF2 (NT1 * KT1 * 512)                   // 214016
#define PK_OFF3 (PK_OFF2 + NT2 * KT2 * 512)         // 249856
#define PK_TOTAL (PK_OFF3 + NT3 * KT3 * 512)        // 251904 bf16 = 503808 B

typedef __attribute__((ext_vector_type(16))) __bf16 v16bf;
typedef __attribute__((ext_vector_type(8)))  float  v8f;
typedef __attribute__((ext_vector_type(4)))  float  f4v;

__device__ __forceinline__ v8f wmma_bf16(v16bf a, v16bf b, v8f c) {
    // D = A(16x32 bf16) * B(32x16 bf16) + C(16x16 f32)
    return __builtin_amdgcn_wmma_f32_16x16x32_bf16(
        /*neg_a=*/false, a, /*neg_b=*/false, b,
        /*c_mod=*/(short)0, c, /*reuse_a=*/false, /*reuse_b=*/false);
}

// ---------------------------------------------------------------------------
// Prepack: f32 row-major W[K][ldw] -> bf16 B-fragments in WMMA lane layout.
// Fragment (nt,ks): lane L holds 16 contiguous bf16:
//   N = nt*16 + (L&15);  K = ks*32 + (i>>3)*16 + (L>>4)*8 + (i&7)   (i = 0..15)
// dst element index = ((nt*ktiles + ks)*32 + L)*16 + i  (coalesced writes)
// ---------------------------------------------------------------------------
__global__ void __launch_bounds__(THREADS)
prepack_b(const float* __restrict__ W, int ldw, int Kvalid, int Nvalid,
          int ktiles, __bf16* __restrict__ dst, int total) {
    int t = blockIdx.x * THREADS + threadIdx.x;
    if (t >= total) return;
    int frag = t >> 9;           // /512
    int r    = t & 511;
    int lane = r >> 4;
    int i    = r & 15;
    int nt   = frag / ktiles;
    int ks   = frag - nt * ktiles;
    int n    = nt * 16 + (lane & 15);
    int k    = ks * 32 + (i >> 3) * 16 + (lane >> 4) * 8 + (i & 7);
    float f  = 0.0f;
    if (k < Kvalid && n < Nvalid) f = W[k * ldw + n];
    dst[t] = (__bf16)f;
}

// A-frag (16x32 bf16) from LDS, rows row0..row0+15, K = k0..k0+31.
__device__ __forceinline__ v16bf load_a_frag(const __bf16* base, int row0, int stride,
                                             int k0, int lane) {
    int m    = lane & 15;
    int half = lane >> 4;
    const __bf16* p = base + (row0 + m) * stride + k0 + half * 8;
    union { uint4 u[2]; v16bf v; } t;
    t.u[0] = *(const uint4*)(p);       // K = k0 + half*8 + {0..7}
    t.u[1] = *(const uint4*)(p + 16);  // K = k0 + 16 + half*8 + {0..7}
    return t.v;
}

// B-frag: two coalesced b128 loads from the prepacked buffer (32 B / lane).
__device__ __forceinline__ v16bf load_b_packed(const __bf16* __restrict__ packed,
                                               int frag, int lane) {
    const uint4* p = (const uint4*)(packed + ((size_t)frag * 32 + lane) * 16);
    union { uint4 u[2]; v16bf v; } t;
    t.u[0] = p[0];
    t.u[1] = p[1];
    return t.v;
}

// Fully-unrolled software-pipelined K-loop: loads for step k+1 are issued
// before the 4 WMMAs of step k. Full unroll (compile-time KT) turns the
// buffer hand-off into SSA renames: no register copies, no WAR-hazard NOPs,
// while keeping a WMMA-group of load slack for relaxed s_wait counts.
template <int KT>
__device__ __forceinline__ void gemm_panel(const __bf16* Abase, int lda,
                                           const __bf16* __restrict__ pk, int fragBase,
                                           v8f acc[4], int lane) {
    v16bf bf = load_b_packed(pk, fragBase, lane);
    v16bf af[4];
#pragma unroll
    for (int mt = 0; mt < 4; ++mt) af[mt] = load_a_frag(Abase, mt * 16, lda, 0, lane);

#pragma unroll
    for (int ks = 0; ks < KT - 1; ++ks) {
        v16bf bn = load_b_packed(pk, fragBase + ks + 1, lane);
        v16bf an[4];
#pragma unroll
        for (int mt = 0; mt < 4; ++mt)
            an[mt] = load_a_frag(Abase, mt * 16, lda, (ks + 1) * 32, lane);
#pragma unroll
        for (int mt = 0; mt < 4; ++mt) acc[mt] = wmma_bf16(af[mt], bf, acc[mt]);
        bf = bn;
#pragma unroll
        for (int mt = 0; mt < 4; ++mt) af[mt] = an[mt];
    }
#pragma unroll
    for (int mt = 0; mt < 4; ++mt) acc[mt] = wmma_bf16(af[mt], bf, acc[mt]);
}

// Store 16x16 f32 C-frag: +bias, ReLU, convert bf16, into LDS [.. x stride].
__device__ __forceinline__ void store_c_relu(__bf16* dst, int row0, int stride, int n0,
                                             v8f acc, const float* __restrict__ bias,
                                             int Nvalid, int lane) {
    int n      = n0 + (lane & 15);
    int half   = lane >> 4;
    int nc     = n < Nvalid ? n : (Nvalid - 1);   // clamped, branch-free load
    float braw = bias[nc];
    float bv   = n < Nvalid ? braw : 0.0f;
#pragma unroll
    for (int r = 0; r < 8; ++r) {
        float v = acc[r] + bv;
        v = v > 0.0f ? v : 0.0f;
        dst[(row0 + half * 8 + r) * stride + n] = (__bf16)v;
    }
}

__global__ void __launch_bounds__(THREADS)
fused_digit_mlp(const float* __restrict__ x,  const float* __restrict__ cw,
                const __bf16* __restrict__ Wp,           // prepacked B frags
                const float* __restrict__ b1, const float* __restrict__ b2,
                const float* __restrict__ b3,
                float* __restrict__ out) {
    __shared__ __align__(16) unsigned char smem[SMEM_BYTES];
    __bf16* feat = (__bf16*)(smem + FEAT_OFF);
    __bf16* h1   = (__bf16*)(smem + H1_OFF);
    __bf16* h2   = (__bf16*)(smem + H2_OFF);
    float*  xs   = (float*)(smem + XS_OFF);

    const int tid     = threadIdx.x;
    const int lane    = tid & 31;
    const int wave    = tid >> 5;
    const int rowBase = blockIdx.x * ROWS_PER_BLK;

    const float c0 = cw[0], c1 = cw[1], c2 = cw[2];
    const float c3 = cw[3], c4 = cw[4], c5 = cw[5];
    const float c6 = cw[6], c7 = cw[7], c8 = cw[8];

    // ---- zero feat (covers K-pad cols 676..703) ----
    {
        unsigned int* p = (unsigned int*)(smem + FEAT_OFF);
        for (int i = tid; i < (ROWS_PER_BLK * 704 * 2) / 4; i += THREADS) p[i] = 0u;
    }
    __syncthreads();

    // ---- conv 3x3 valid, 4 chunks of 16 rows staged through LDS ----
    // x is a single-use 205 MB stream: non-temporal loads keep L2 for weights.
    for (int c = 0; c < 4; ++c) {
        const f4v* xg = (const f4v*)(x + (size_t)(rowBase + c * 16) * (IMGD * IMGD));
        f4v* xs4 = (f4v*)xs;
        for (int i = tid; i < (16 * IMGD * IMGD) / 4; i += THREADS)
            xs4[i] = __builtin_nontemporal_load(&xg[i]);
        if (c < 3) {  // prefetch next chunk (global_prefetch_b8)
            const float* xn = x + (size_t)(rowBase + (c + 1) * 16) * (IMGD * IMGD);
            for (int i = tid * 32; i < 16 * IMGD * IMGD; i += THREADS * 32)
                __builtin_prefetch(xn + i, 0, 1);
        }
        __syncthreads();
        for (int idx = tid; idx < 16 * CONV_OUT; idx += THREADS) {
            int r  = idx / CONV_OUT;
            int p  = idx - r * CONV_OUT;
            int oy = p / 26;
            int ox = p - oy * 26;
            const float* s = xs + r * (IMGD * IMGD) + oy * IMGD + ox;
            float v = s[0]  * c0 + s[1]  * c1 + s[2]  * c2
                    + s[28] * c3 + s[29] * c4 + s[30] * c5
                    + s[56] * c6 + s[57] * c7 + s[58] * c8;
            feat[(c * 16 + r) * 704 + p] = (__bf16)v;
        }
        __syncthreads();
    }

    // ---- zero h1 + h2 regions (x-stage is dead now) ----
    {
        unsigned int* p = (unsigned int*)(smem + H1_OFF);
        for (int i = tid; i < (SMEM_BYTES - H1_OFF) / 4; i += THREADS) p[i] = 0u;
    }
    __syncthreads();

    // ---- GEMM1: h1[64,300] = relu(feat[64,676] @ W1[676,300] + b1) ----
    for (int nt = wave; nt < NT1; nt += 8) {
        v8f acc[4] = {};
        gemm_panel<KT1>(feat, 704, Wp + PK_OFF1, nt * KT1, acc, lane);
#pragma unroll
        for (int mt = 0; mt < 4; ++mt)
            store_c_relu(h1, mt * 16, 320, nt * 16, acc[mt], b1, 300, lane);
    }
    __syncthreads();

    // ---- GEMM2: h2[64,100] = relu(h1[64,300] @ W2[300,100] + b2) ----
    for (int nt = wave; nt < NT2; nt += 8) {
        v8f acc[4] = {};
        gemm_panel<KT2>(h1, 320, Wp + PK_OFF2, nt * KT2, acc, lane);
#pragma unroll
        for (int mt = 0; mt < 4; ++mt)
            store_c_relu(h2, mt * 16, 128, nt * 16, acc[mt], b2, 100, lane);
    }
    __syncthreads();

    // ---- GEMM3: out[64,10] = h2[64,100] @ W3[100,10] + b3 ----
    if (wave < 4) {
        const __bf16* pk = Wp + PK_OFF3;
        const int mt = wave;
        v8f acc = {};
#pragma unroll
        for (int ks = 0; ks < KT3; ++ks) {
            v16bf bf = load_b_packed(pk, ks, lane);
            v16bf af = load_a_frag(h2, mt * 16, 128, ks * 32, lane);
            acc = wmma_bf16(af, bf, acc);
        }
        int n    = lane & 15;
        int half = lane >> 4;
        if (n < 10) {
            float bv = b3[n];
#pragma unroll
            for (int r = 0; r < 8; ++r) {
                int row = rowBase + mt * 16 + half * 8 + r;
                __builtin_nontemporal_store(acc[r] + bv, &out[(size_t)row * 10 + n]);
            }
        }
    }
}

extern "C" void kernel_launch(void* const* d_in, const int* in_sizes, int n_in,
                              void* d_out, int out_size, void* d_ws, size_t ws_size,
                              hipStream_t stream) {
    const float* x  = (const float*)d_in[0];
    const float* cw = (const float*)d_in[1];
    const float* W1 = (const float*)d_in[2];
    const float* b1 = (const float*)d_in[3];
    const float* W2 = (const float*)d_in[4];
    const float* b2 = (const float*)d_in[5];
    const float* W3 = (const float*)d_in[6];
    const float* b3 = (const float*)d_in[7];
    float* out = (float*)d_out;
    __bf16* Wp = (__bf16*)d_ws;   // PK_TOTAL bf16 = 503,808 bytes of scratch
    (void)in_sizes; (void)n_in; (void)out_size; (void)ws_size;

    // Pre-pack weights into WMMA B-fragment bf16 layout (every call; deterministic).
    {
        int tot1 = NT1 * KT1 * 512;
        int tot2 = NT2 * KT2 * 512;
        int tot3 = NT3 * KT3 * 512;
        hipLaunchKernelGGL(prepack_b, dim3((tot1 + THREADS - 1) / THREADS), dim3(THREADS),
                           0, stream, W1, 300, 676, 300, KT1, Wp + PK_OFF1, tot1);
        hipLaunchKernelGGL(prepack_b, dim3((tot2 + THREADS - 1) / THREADS), dim3(THREADS),
                           0, stream, W2, 100, 300, 100, KT2, Wp + PK_OFF2, tot2);
        hipLaunchKernelGGL(prepack_b, dim3((tot3 + THREADS - 1) / THREADS), dim3(THREADS),
                           0, stream, W3, 10, 100, 10, KT3, Wp + PK_OFF3, tot3);
    }

    dim3 grid(BATCH / ROWS_PER_BLK);
    dim3 block(THREADS);
    hipLaunchKernelGGL(fused_digit_mlp, grid, block, 0, stream,
                       x, cw, Wp, b1, b2, b3, out);
}
